// LocalSelfAttentionBlock_70909910057566
// MI455X (gfx1250) — matrix operand order repaired and checked by reading, not compiled, not run on hardware
//
#include <hip/hip_runtime.h>
#include <stdint.h>

// ---------------------------------------------------------------- constants
constexpr int CB    = 2;      // batch
constexpr int CS    = 4096;   // sequence
constexpr int CD    = 512;    // model dim
constexpr int CH    = 8;      // heads
constexpr int CHD   = 64;     // head dim
constexpr int CW    = 512;    // window / chunk
constexpr int CNC   = 8;      // chunks
constexpr int CHALF = 256;    // half window

#define LDT 40      // padded LDS row stride (elements) for GEMM tiles
#define KSTEPS 16   // 512 / 32

typedef __attribute__((ext_vector_type(16))) __bf16    v16bf;
typedef __attribute__((ext_vector_type(8)))  float     v8f;
typedef __attribute__((ext_vector_type(8)))  unsigned  v8u;
typedef __attribute__((ext_vector_type(4)))  unsigned  v4u;
typedef __attribute__((ext_vector_type(4)))  int       i4v;

// ------------------------------------------------- async global->LDS support
#if defined(__has_builtin)
#if __has_builtin(__builtin_amdgcn_global_load_async_to_lds_b128)
#define USE_ASYNC_LDS 1
#endif
#endif
#ifndef USE_ASYNC_LDS
#define USE_ASYNC_LDS 0
#endif

// copy 16 bytes (8 bf16) global -> LDS, per lane
__device__ __forceinline__ void async_copy16(unsigned short* ldst,
                                             const unsigned short* gsrc) {
#if USE_ASYNC_LDS
  __builtin_amdgcn_global_load_async_to_lds_b128(
      (__attribute__((address_space(1))) i4v*)gsrc,
      (__attribute__((address_space(3))) i4v*)ldst, 0, 0);
#else
  *(uint4*)ldst = *(const uint4*)gsrc;
#endif
}

__device__ __forceinline__ void async_wait0() {
#if USE_ASYNC_LDS
#if __has_builtin(__builtin_amdgcn_s_wait_asynccnt)
  __builtin_amdgcn_s_wait_asynccnt(0);
#else
  asm volatile("s_wait_asynccnt 0x0" ::: "memory");
#endif
#endif
}

// ---------------------------------------------------------------- helpers
__device__ __forceinline__ unsigned short f2bf(float f) {
  unsigned int u = __float_as_uint(f);
  u = (u + 0x7FFFu + ((u >> 16) & 1u)) >> 16;
  return (unsigned short)u;
}

// Build a 16x32 (A) or 32x16 (B) bf16 WMMA fragment from two 16-byte chunks
// (VGPRs 0-3 from p0, VGPRs 4-7 from p1), per the CDNA5 16-bit layouts.
__device__ __forceinline__ v16bf make_frag(const unsigned short* p0,
                                           const unsigned short* p1) {
  v4u lo = *(const v4u*)p0;
  v4u hi = *(const v4u*)p1;
  union { v8u u; v16bf b; } f;
  f.u[0] = lo[0]; f.u[1] = lo[1]; f.u[2] = lo[2]; f.u[3] = lo[3];
  f.u[4] = hi[0]; f.u[5] = hi[1]; f.u[6] = hi[2]; f.u[7] = hi[3];
  return f.b;
}

__device__ __forceinline__ v8f wmma_bf16(v16bf a, v16bf b, v8f c) {
  return __builtin_amdgcn_wmma_f32_16x16x32_bf16(false, a, false, b, (short)0,
                                                 c, false, false);
}

// ---------------------------------------------------------------- GEMM core
// One 128-thread block (4 waves) computes a 128(M) x 64(N) tile of
// C = A[MxK] * B[KxN]; each wave owns 32 M-rows (two 16-row fragments).
// Double-buffered LDS; A staged by async global->LDS b128 copies, B staged
// transposed ([n][k]) through VGPRs so B fragments are contiguous b128 loads.
__device__ __forceinline__ void gemm_core_128x64(
    const unsigned short* __restrict__ Ag, int lda,
    const unsigned short* __restrict__ Bg, int ldb,
    unsigned short* lA0, unsigned short* lA1,
    unsigned short* lB0, unsigned short* lB1, v8f acc[2][4]) {
  const int tid  = threadIdx.x;
  const int wave = tid >> 5;
  const int lane = tid & 31;
  const int m    = lane & 15;
  const int half = lane >> 4;
  const int ar   = tid;               // A staging: one 32-elem row per thread
  const int bk   = tid & 31;          // B staging coords
  const int bn0  = (tid >> 5) << 4;

  // ---- prologue: stage k-step 0 into buffer 0
  {
    const unsigned short* as = Ag + (size_t)ar * lda;
#pragma unroll
    for (int j = 0; j < 4; ++j) async_copy16(lA0 + ar * LDT + j * 8, as + j * 8);
    const uint4* bs = (const uint4*)(Bg + (size_t)bk * ldb + bn0);
    uint4 b0 = bs[0], b1 = bs[1];
    unsigned int w[8] = {b0.x, b0.y, b0.z, b0.w, b1.x, b1.y, b1.z, b1.w};
#pragma unroll
    for (int j = 0; j < 8; ++j) {
      lB0[(bn0 + 2 * j)     * LDT + bk] = (unsigned short)(w[j] & 0xffffu);
      lB0[(bn0 + 2 * j + 1) * LDT + bk] = (unsigned short)(w[j] >> 16);
    }
  }
  async_wait0();
  __syncthreads();

  for (int i = 0; i < KSTEPS; ++i) {
    unsigned short* curA = (i & 1) ? lA1 : lA0;
    unsigned short* curB = (i & 1) ? lB1 : lB0;
    unsigned short* nxtA = (i & 1) ? lA0 : lA1;
    unsigned short* nxtB = (i & 1) ? lB0 : lB1;
    const bool has_next = (i + 1 < KSTEPS);
    uint4 b0, b1;
    b0.x = b0.y = b0.z = b0.w = 0u;
    b1 = b0;
    if (has_next) {  // issue next-step staging before math (latency hiding)
      const int kk = (i + 1) * 32;
      const unsigned short* as = Ag + (size_t)ar * lda + kk;
#pragma unroll
      for (int j = 0; j < 4; ++j)
        async_copy16(nxtA + ar * LDT + j * 8, as + j * 8);
      const uint4* bs = (const uint4*)(Bg + (size_t)(kk + bk) * ldb + bn0);
      b0 = bs[0];
      b1 = bs[1];
    }
    // ---- compute on current buffers
    const unsigned short* arow0 = curA + (wave * 32 + m) * LDT;
    const unsigned short* arow1 = curA + (wave * 32 + 16 + m) * LDT;
    v16bf af0 = make_frag(arow0 + half * 8, arow0 + 16 + half * 8);
    v16bf af1 = make_frag(arow1 + half * 8, arow1 + 16 + half * 8);
#pragma unroll
    for (int nt = 0; nt < 4; ++nt) {
      const unsigned short* bcol = curB + (nt * 16 + m) * LDT;
      v16bf bfr = make_frag(bcol + half * 8, bcol + 16 + half * 8);
      acc[0][nt] = wmma_bf16(af0, bfr, acc[0][nt]);
      acc[1][nt] = wmma_bf16(af1, bfr, acc[1][nt]);
    }
    if (has_next) {  // finish B staging (transpose through VGPRs)
      unsigned int w[8] = {b0.x, b0.y, b0.z, b0.w, b1.x, b1.y, b1.z, b1.w};
#pragma unroll
      for (int j = 0; j < 8; ++j) {
        nxtB[(bn0 + 2 * j)     * LDT + bk] = (unsigned short)(w[j] & 0xffffu);
        nxtB[(bn0 + 2 * j + 1) * LDT + bk] = (unsigned short)(w[j] >> 16);
      }
    }
    async_wait0();
    __syncthreads();
  }
}

// ---------------------------------------------------------------- kernels
// fp32 -> bf16 conversion of activations + weights (QKV weights fused [512][1536])
__global__ __launch_bounds__(256) void convert_kernel(
    const float* __restrict__ val, const float* __restrict__ Wq,
    const float* __restrict__ Wk, const float* __restrict__ Wv,
    const float* __restrict__ Wo, unsigned short* __restrict__ xb,
    unsigned short* __restrict__ wqkv, unsigned short* __restrict__ wob) {
  size_t i = (size_t)blockIdx.x * 256 + threadIdx.x;
  const size_t NV = (size_t)CB * CS * CD;   // 4194304
  const size_t NW = (size_t)CD * CD;        // 262144
  if (i < NV) { xb[i] = f2bf(val[i]); return; }
  i -= NV;
  if (i < 3 * NW) {
    const int mat = (int)(i / NW);
    const size_t j = i % NW;
    const int k = (int)(j >> 9), n = (int)(j & 511);
    const float* w = (mat == 0) ? Wq : (mat == 1 ? Wk : Wv);
    wqkv[(size_t)k * (3 * CD) + mat * CD + n] = f2bf(w[j]);
    return;
  }
  i -= 3 * NW;
  if (i < NW) wob[i] = f2bf(Wo[i]);
}

// Fused QKV projection: [8192x512] x [512x1536] -> per-head bf16 q/k/v.
__global__ __launch_bounds__(128) void qkv_gemm_kernel(
    const unsigned short* __restrict__ xb, const unsigned short* __restrict__ wqkv,
    const float* __restrict__ bq, const float* __restrict__ bk,
    const float* __restrict__ bv, unsigned short* __restrict__ qb,
    unsigned short* __restrict__ kb, unsigned short* __restrict__ vb) {
  __shared__ __align__(16) unsigned short lA0[128 * LDT];
  __shared__ __align__(16) unsigned short lA1[128 * LDT];
  __shared__ __align__(16) unsigned short lB0[64 * LDT];
  __shared__ __align__(16) unsigned short lB1[64 * LDT];
  const int mtile = blockIdx.x;   // 0..63
  const int ntile = blockIdx.y;   // 0..23
  v8f acc[2][4];
#pragma unroll
  for (int r = 0; r < 2; ++r)
#pragma unroll
    for (int nt = 0; nt < 4; ++nt)
#pragma unroll
      for (int v = 0; v < 8; ++v) acc[r][nt][v] = 0.0f;

  gemm_core_128x64(xb + (size_t)mtile * 128 * CD, CD, wqkv + ntile * 64, 3 * CD,
                   lA0, lA1, lB0, lB1, acc);

  const int lane = threadIdx.x & 31, wave = threadIdx.x >> 5;
  const int half = lane >> 4;
  const int mat  = ntile >> 3;         // 0=q 1=k 2=v
  const int h    = ntile & 7;          // head
  const float* bias = (mat == 0) ? bq : (mat == 1 ? bk : bv);
  unsigned short* dst = (mat == 0) ? qb : (mat == 1 ? kb : vb);
#pragma unroll
  for (int r = 0; r < 2; ++r)
#pragma unroll
    for (int nt = 0; nt < 4; ++nt) {
      const int hd = nt * 16 + (lane & 15);
      const float bval = bias[h * CHD + hd];
#pragma unroll
      for (int v = 0; v < 8; ++v) {
        const int m = mtile * 128 + wave * 32 + r * 16 + v + 8 * half;
        const int b = m >> 12;           // / 4096
        const int s = m & (CS - 1);
        dst[((size_t)(b * CH + h) * CS + s) * CHD + hd] = f2bf(acc[r][nt][v] + bval);
      }
    }
}

// Banded flash attention over one (b,h,chunk,64-query-rows) tile.
__global__ __launch_bounds__(128) void attn_kernel(
    const unsigned short* __restrict__ qb, const unsigned short* __restrict__ kb,
    const unsigned short* __restrict__ vb, unsigned short* __restrict__ ao) {
  __shared__ __align__(16) unsigned short lK[64 * 72];   // [key][hd]
  __shared__ __align__(16) unsigned short lV[64 * 72];   // [hd][key] (transposed)
  __shared__ __align__(16) unsigned short lP[4][16 * 32];

  const int bid   = blockIdx.x;          // (b*H+h)*NC + c
  const int c     = bid & (CNC - 1);
  const int bh    = bid >> 3;
  const int qbase = blockIdx.y * 64;

  const int tid  = threadIdx.x;
  const int wave = tid >> 5;
  const int lane = tid & 31;
  const int l15  = lane & 15;
  const int half = lane >> 4;

  const unsigned short* qchunk = qb + ((size_t)bh * CS + c * CW) * CHD;

  v16bf qf0, qf1;
  {
    const unsigned short* qrow = qchunk + (size_t)(qbase + wave * 16 + l15) * CHD;
    qf0 = make_frag(qrow + half * 8,      qrow + 16 + half * 8);
    qf1 = make_frag(qrow + 32 + half * 8, qrow + 48 + half * 8);
  }

  float mrow[8], lrow[8];
  v8f oacc[4];
#pragma unroll
  for (int v = 0; v < 8; ++v) { mrow[v] = -3.0e38f; lrow[v] = 0.0f; }
#pragma unroll
  for (int nt = 0; nt < 4; ++nt)
#pragma unroll
    for (int v = 0; v < 8; ++v) oacc[nt][v] = 0.0f;

  unsigned short* lPw = &lP[wave][0];
  const float scale = 0.125f;   // 1/sqrt(HD)

  for (int t = 0; t < 9; ++t) {
    const int kj0   = qbase + CHALF + t * 64;   // first key index in [0,3W)
    const int spos0 = (c - 1) * CW + kj0;       // its global sequence position
    __syncthreads();
    {  // stage K (async, straight) and V (transposed); OOB keys -> zeros
      const int r  = tid >> 1;
      const int h0 = (tid & 1) << 5;
      const int sp = spos0 + r;
      // prefetch next tile's K/V rows (clamped address, speculative)
      {
        int spn = sp + 64;
        spn = spn < 0 ? 0 : (spn >= CS ? CS - 1 : spn);
        const size_t gofs = ((size_t)bh * CS + spn) * CHD + h0;
        __builtin_prefetch(kb + gofs, 0, 0);
        __builtin_prefetch(vb + gofs, 0, 0);
      }
      if (sp >= 0 && sp < CS) {
        const unsigned short* gk = kb + ((size_t)bh * CS + sp) * CHD + h0;
#pragma unroll
        for (int j = 0; j < 4; ++j)
          async_copy16(lK + r * 72 + h0 + j * 8, gk + j * 8);
        const uint4* sv = (const uint4*)(vb + ((size_t)bh * CS + sp) * CHD + h0);
        uint4 dv0 = sv[0], dv1 = sv[1], dv2 = sv[2], dv3 = sv[3];
        unsigned int w[16] = {dv0.x, dv0.y, dv0.z, dv0.w, dv1.x, dv1.y, dv1.z,
                              dv1.w, dv2.x, dv2.y, dv2.z, dv2.w, dv3.x, dv3.y,
                              dv3.z, dv3.w};
#pragma unroll
        for (int i = 0; i < 16; ++i) {
          lV[(h0 + 2 * i)     * 72 + r] = (unsigned short)(w[i] & 0xffffu);
          lV[(h0 + 2 * i + 1) * 72 + r] = (unsigned short)(w[i] >> 16);
        }
      } else {
        uint4 z; z.x = z.y = z.z = z.w = 0u;
        uint4* dk = (uint4*)(lK + r * 72 + h0);
        dk[0] = z; dk[1] = z; dk[2] = z; dk[3] = z;
#pragma unroll
        for (int i = 0; i < 32; ++i) lV[(h0 + i) * 72 + r] = 0;
      }
    }
    async_wait0();
    __syncthreads();

#pragma unroll
    for (int g = 0; g < 2; ++g) {
#pragma unroll
      for (int sub = 0; sub < 2; ++sub) {
        const int ksub = g * 32 + sub * 16;
        v8f sacc;
#pragma unroll
        for (int v = 0; v < 8; ++v) sacc[v] = 0.0f;
        {
          const unsigned short* krow = lK + (size_t)(ksub + l15) * 72;
          v16bf kf0 = make_frag(krow + half * 8,      krow + 16 + half * 8);
          v16bf kf1 = make_frag(krow + 32 + half * 8, krow + 48 + half * 8);
          sacc = wmma_bf16(qf0, kf0, sacc);
          sacc = wmma_bf16(qf1, kf1, sacc);
        }
        // mask + online softmax (C layout: lane = column, acc idx = row)
        const int kj = kj0 + ksub + l15;
        const int sp = spos0 + ksub + l15;
        const bool keyok = (sp >= 0) && (sp < CS);
        float corr[8], p[8];
#pragma unroll
        for (int v = 0; v < 8; ++v) {
          const int qi  = qbase + wave * 16 + v + 8 * half;
          const int rel = kj - CW - qi;
          const bool ok = keyok && (rel >= -CHALF) && (rel <= CHALF);
          float sc = ok ? sacc[v] * scale : -3.0e38f;
          float mx = sc;
          mx = fmaxf(mx, __shfl_xor(mx, 1, 32));
          mx = fmaxf(mx, __shfl_xor(mx, 2, 32));
          mx = fmaxf(mx, __shfl_xor(mx, 4, 32));
          mx = fmaxf(mx, __shfl_xor(mx, 8, 32));
          const float mnew = fmaxf(mrow[v], mx);
          const float pv = ok ? __expf(sc - mnew) : 0.0f;
          float rs = pv;
          rs += __shfl_xor(rs, 1, 32);
          rs += __shfl_xor(rs, 2, 32);
          rs += __shfl_xor(rs, 4, 32);
          rs += __shfl_xor(rs, 8, 32);
          corr[v] = __expf(mrow[v] - mnew);
          lrow[v] = lrow[v] * corr[v] + rs;
          mrow[v] = mnew;
          p[v]    = pv;
        }
#pragma unroll
        for (int nt = 0; nt < 4; ++nt)
#pragma unroll
          for (int v = 0; v < 8; ++v) oacc[nt][v] *= corr[v];
#pragma unroll
        for (int v = 0; v < 8; ++v)
          lPw[(v + 8 * half) * 32 + sub * 16 + l15] = f2bf(p[v]);
      }
      // O += P(16x32) x V(32x64); per-wave LDS P, DS ops in-order per wave
      {
        const unsigned short* prow = lPw + (size_t)l15 * 32;
        v16bf pf = make_frag(prow + half * 8, prow + 16 + half * 8);
#pragma unroll
        for (int nt = 0; nt < 4; ++nt) {
          const unsigned short* vcol =
              lV + (size_t)(nt * 16 + l15) * 72 + g * 32;
          v16bf vf = make_frag(vcol + half * 8, vcol + 16 + half * 8);
          oacc[nt] = wmma_bf16(pf, vf, oacc[nt]);
        }
      }
    }
  }

  const int b = bh >> 3, h = bh & 7;
  float inv[8];
#pragma unroll
  for (int v = 0; v < 8; ++v) inv[v] = 1.0f / lrow[v];
#pragma unroll
  for (int nt = 0; nt < 4; ++nt) {
    const int hd = nt * 16 + l15;
#pragma unroll
    for (int v = 0; v < 8; ++v) {
      const int qi = qbase + wave * 16 + v + 8 * half;
      const int s  = c * CW + qi;
      ao[((size_t)b * CS + s) * CD + h * CHD + hd] = f2bf(oacc[nt][v] * inv[v]);
    }
  }
}

// Output projection + bias + residual: proj = ao @ Wo + bo + val (fp32)
__global__ __launch_bounds__(128) void proj_gemm_kernel(
    const unsigned short* __restrict__ ao, const unsigned short* __restrict__ wob,
    const float* __restrict__ bo, const float* __restrict__ val,
    float* __restrict__ proj) {
  __shared__ __align__(16) unsigned short lA0[128 * LDT];
  __shared__ __align__(16) unsigned short lA1[128 * LDT];
  __shared__ __align__(16) unsigned short lB0[64 * LDT];
  __shared__ __align__(16) unsigned short lB1[64 * LDT];
  const int mtile = blockIdx.x;   // 0..63
  const int ntile = blockIdx.y;   // 0..7
  v8f acc[2][4];
#pragma unroll
  for (int r = 0; r < 2; ++r)
#pragma unroll
    for (int nt = 0; nt < 4; ++nt)
#pragma unroll
      for (int v = 0; v < 8; ++v) acc[r][nt][v] = 0.0f;

  gemm_core_128x64(ao + (size_t)mtile * 128 * CD, CD, wob + ntile * 64, CD,
                   lA0, lA1, lB0, lB1, acc);

  const int lane = threadIdx.x & 31, wave = threadIdx.x >> 5;
  const int half = lane >> 4;
#pragma unroll
  for (int r = 0; r < 2; ++r)
#pragma unroll
    for (int nt = 0; nt < 4; ++nt) {
      const int n = ntile * 64 + nt * 16 + (lane & 15);
      const float bval = bo[n];
#pragma unroll
      for (int v = 0; v < 8; ++v) {
        const int m = mtile * 128 + wave * 32 + r * 16 + v + 8 * half;
        const size_t idx = (size_t)m * CD + n;
        proj[idx] = acc[r][nt][v] + bval + val[idx];
      }
    }
}

// Row-wise LayerNorm over D=512
__device__ __forceinline__ float block_sum_256(float v, float* red) {
  const int lane = threadIdx.x & 31, wv = threadIdx.x >> 5;
#pragma unroll
  for (int o = 16; o > 0; o >>= 1) v += __shfl_xor(v, o, 32);
  __syncthreads();
  if (lane == 0) red[wv] = v;
  __syncthreads();
  float t = red[0];
#pragma unroll
  for (int i = 1; i < 8; ++i) t += red[i];
  return t;
}

__global__ __launch_bounds__(256) void layernorm_kernel(
    const float* __restrict__ x, const float* __restrict__ gamma,
    const float* __restrict__ beta, float* __restrict__ out) {
  __shared__ float red[8];
  const int row = blockIdx.x;
  const float* xr = x + (size_t)row * CD;
  const int t = threadIdx.x;
  const float a0 = xr[t], a1 = xr[t + 256];
  const float s  = block_sum_256(a0 + a1, red);
  const float mu = s * (1.0f / 512.0f);
  const float d0 = a0 - mu, d1 = a1 - mu;
  const float s2 = block_sum_256(d0 * d0 + d1 * d1, red);
  const float r  = rsqrtf(s2 * (1.0f / 512.0f) + 1e-5f);
  out[(size_t)row * CD + t]       = d0 * r * gamma[t] + beta[t];
  out[(size_t)row * CD + t + 256] = d1 * r * gamma[t + 256] + beta[t + 256];
}

// ---------------------------------------------------------------- launch
extern "C" void kernel_launch(void* const* d_in, const int* in_sizes, int n_in,
                              void* d_out, int out_size, void* d_ws,
                              size_t ws_size, hipStream_t stream) {
  const float* val   = (const float*)d_in[0];
  const float* Wq    = (const float*)d_in[1];
  const float* bq    = (const float*)d_in[2];
  const float* Wk    = (const float*)d_in[3];
  const float* bk    = (const float*)d_in[4];
  const float* Wv    = (const float*)d_in[5];
  const float* bv    = (const float*)d_in[6];
  const float* Wo    = (const float*)d_in[7];
  const float* bo    = (const float*)d_in[8];
  const float* gamma = (const float*)d_in[9];
  const float* beta  = (const float*)d_in[10];
  float* out = (float*)d_out;

  char* ws = (char*)d_ws;
  size_t off = 0;
  auto carve = [&](size_t bytes) -> char* {
    char* p = ws + off;
    off += (bytes + 255) & ~(size_t)255;
    return p;
  };
  const size_t NV = (size_t)CB * CS * CD;   // 4194304 elements
  const size_t NW = (size_t)CD * CD;
  unsigned short* xb   = (unsigned short*)carve(NV * 2);
  unsigned short* wqkv = (unsigned short*)carve(3 * NW * 2);
  unsigned short* wob  = (unsigned short*)carve(NW * 2);
  unsigned short* qb   = (unsigned short*)carve(NV * 2);
  unsigned short* kb   = (unsigned short*)carve(NV * 2);
  unsigned short* vb   = (unsigned short*)carve(NV * 2);
  unsigned short* ao   = (unsigned short*)carve(NV * 2);
  float*          proj = (float*)carve(NV * 4);

  {  // fp32 -> bf16 conversions
    const size_t total = NV + 4 * NW;
    dim3 grid((unsigned)((total + 255) / 256));
    convert_kernel<<<grid, 256, 0, stream>>>(val, Wq, Wk, Wv, Wo, xb, wqkv, wob);
  }
  {  // fused QKV projection
    dim3 grid(64, 24);
    qkv_gemm_kernel<<<grid, 128, 0, stream>>>(xb, wqkv, bq, bk, bv, qb, kb, vb);
  }
  {  // banded attention
    dim3 grid(CB * CH * CNC, CW / 64);
    attn_kernel<<<grid, 128, 0, stream>>>(qb, kb, vb, ao);
  }
  {  // output projection + residual
    dim3 grid(64, 8);
    proj_gemm_kernel<<<grid, 128, 0, stream>>>(ao, wob, bo, val, proj);
  }
  {  // layer norm
    layernorm_kernel<<<CB * CS, 256, 0, stream>>>(proj, gamma, beta, out);
  }
}
